// NewSchNetWrap_5059471475333
// MI455X (gfx1250) — matrix operand order repaired and checked by reading, not compiled, not run on hardware
//
#include <hip/hip_runtime.h>

typedef __attribute__((ext_vector_type(16))) _Float16 v16h;
typedef __attribute__((ext_vector_type(8)))  float    v8f;

#define N_NODES   20000
#define N_EDGES   640000
#define N_GRAPHS  100
#define HID       128
#define NFILT     128
#define NG        50
#define NGP       64      // gaussians padded to multiple of 32 for WMMA K
#define NI        6
#define ET        32      // edge rows per block (2 M-tiles)
#define NT        32      // node rows per block (2 M-tiles)

#define PI_F      3.14159265358979f
#define CUTOFF_F  10.0f
#define GSTEP     (CUTOFF_F / (float)(NG - 1))
#define GCOEFF    (-0.5f / (GSTEP * GSTEP))
#define LOG2_F    0.6931471805599453f

__device__ __forceinline__ float sspf(float x) {
  // shifted softplus: log(1+exp(x)) - log(2)
  float sp = (x > 20.0f) ? x : __logf(1.0f + __expf(x));
  return sp - LOG2_F;
}

// ---------------- WMMA fragment helpers (A/B k = 16*hf + j, not ISA table order; D per CDNA5 ISA)

// A (16x32 f16 tile) from row-major LDS buffer As[rows][ld]; row = M row, hf = lane>>4
__device__ __forceinline__ v16h load_a_lds(const _Float16* As, int ld, int row, int hf, int k0) {
  v16h a;
  const _Float16* p = As + row * ld + k0 + hf * 16;
#pragma unroll
  for (int j = 0; j < 8; ++j) a[j] = p[j];          // K = k0 + hf*16 + j
#pragma unroll
  for (int j = 0; j < 8; ++j) a[8 + j] = p[8 + j];  // K = k0 + hf*16 + 8 + j
  return a;
}

// Fragment-ready weight layout: element (k,n) of a [K][128] matrix lives at
//   (k>>5)*4096 + (n>>4)*512 + lane*16 + j,  lane = ((k>>4)&1)*16 + (n&15), j = k&15
// so each lane's 16-half B fragment is one contiguous 32-byte block.
__device__ __forceinline__ int frag_idx(int k, int n) {
  return ((k >> 5) << 12) + ((n >> 4) << 9) + (((((k >> 4) & 1) << 4) + (n & 15)) << 4) + (k & 15);
}

__device__ __forceinline__ v16h load_b_frag(const _Float16* W, int kb, int wave, int lane) {
  return *(const v16h*)(W + (kb << 12) + (wave << 9) + (lane << 4));
}

// ------------------------------- utility kernels -------------------------------

__global__ void zero_f32_kernel(float* __restrict__ p, int n) {
  int i = blockIdx.x * blockDim.x + threadIdx.x;
  for (; i < n; i += gridDim.x * blockDim.x) p[i] = 0.0f;
}

__global__ void convert_weights_kernel(const float* __restrict__ m1, const float* __restrict__ m2,
                                       const float* __restrict__ c1, const float* __restrict__ c2,
                                       const float* __restrict__ l,
                                       _Float16* __restrict__ o1, _Float16* __restrict__ o2,
                                       _Float16* __restrict__ oc1, _Float16* __restrict__ oc2,
                                       _Float16* __restrict__ ol) {
  const int S1 = NI * NGP * NFILT;     // padded mlp1 (64x128 per interaction)
  const int S  = NI * NFILT * NFILT;   // square mats
  int tid = blockIdx.x * blockDim.x + threadIdx.x;
  int total = S1 + 4 * S;
  for (; tid < total; tid += gridDim.x * blockDim.x) {
    if (tid < S1) {
      int i = tid / (NGP * NFILT);
      int rem = tid % (NGP * NFILT);
      int k = rem / NFILT, n = rem % NFILT;
      float v = (k < NG) ? m1[((size_t)i * NG + k) * NFILT + n] : 0.0f;
      o1[i * (NGP * NFILT) + frag_idx(k, n)] = (_Float16)v;
    } else {
      int t2 = tid - S1;
      int which = t2 / S;
      int r = t2 % S;
      int i = r / (NFILT * NFILT);
      int rem = r % (NFILT * NFILT);
      int k = rem / NFILT, n = rem % NFILT;
      int o = i * (NFILT * NFILT) + frag_idx(k, n);
      if      (which == 0) o2 [o] = (_Float16)m2[r];
      else if (which == 1) oc1[o] = (_Float16)c1[r];
      else if (which == 2) oc2[o] = (_Float16)c2[r];
      else                 ol [o] = (_Float16)l [r];
    }
  }
}

__global__ void embed_kernel(const int* __restrict__ z, const float* __restrict__ emb,
                             float* __restrict__ h) {
  int tid = blockIdx.x * blockDim.x + threadIdx.x;
  int total = N_NODES * HID;
  for (; tid < total; tid += gridDim.x * blockDim.x) {
    int n = tid >> 7, c = tid & 127;
    h[tid] = emb[(size_t)z[n] * HID + c];
  }
}

// stage 32 contiguous f32 rows [NT x 128] into f16 LDS via float4 loads
__device__ __forceinline__ void stage_rows_f16(const float* __restrict__ g, _Float16* __restrict__ s, int t) {
  const float4* g4 = (const float4*)g;
  for (int idx = t; idx < NT * HID / 4; idx += 256) {
    float4 v = g4[idx];
    _Float16* d = s + idx * 4;
    d[0] = (_Float16)v.x; d[1] = (_Float16)v.y; d[2] = (_Float16)v.z; d[3] = (_Float16)v.w;
  }
}

// ------------------------------- node GEMM: xf = h @ W_cl1 -------------------------------

__global__ __launch_bounds__(256) void lin1_kernel(const float* __restrict__ h,
                                                   const _Float16* __restrict__ W16,
                                                   float* __restrict__ xf) {
  __shared__ _Float16 As[NT][HID];
  int n0 = blockIdx.x * NT;
  int t = threadIdx.x;
  stage_rows_f16(h + (size_t)n0 * HID, &As[0][0], t);
  __syncthreads();

  int wave = t >> 5, lane = t & 31;
  int hf = lane >> 4, lr = lane & 15;
  int ncol = wave * 16 + lr;

  v8f acc0 = {}, acc1 = {};
  for (int kb = 0; kb < HID / 32; ++kb) {
    v16h b  = load_b_frag(W16, kb, wave, lane);
    v16h a0 = load_a_lds(&As[0][0], HID, lr,      hf, kb * 32);
    v16h a1 = load_a_lds(&As[0][0], HID, 16 + lr, hf, kb * 32);
    acc0 = __builtin_amdgcn_wmma_f32_16x16x32_f16(false, a0, false, b, (short)0, acc0, false, false);
    acc1 = __builtin_amdgcn_wmma_f32_16x16x32_f16(false, a1, false, b, (short)0, acc1, false, false);
  }
#pragma unroll
  for (int v = 0; v < 8; ++v) {
    int M = v + 8 * hf;
    xf[(size_t)(n0 + M) * NFILT + ncol]      = acc0[v];
    xf[(size_t)(n0 + 16 + M) * NFILT + ncol] = acc1[v];
  }
}

// ------- fused edge kernel: gaussians -> MLP(WMMA x2) -> *C -> gather*filter -> scatter-add -------

__global__ __launch_bounds__(256) void edge_kernel(const float* __restrict__ pos,
                                                   const int* __restrict__ src, const int* __restrict__ dst,
                                                   const _Float16* __restrict__ Wm1, const float* __restrict__ bm1,
                                                   const _Float16* __restrict__ Wm2, const float* __restrict__ bm2,
                                                   const float* __restrict__ xf, float* __restrict__ agg) {
  __shared__ _Float16 As[ET][NGP];     // padded gaussian features (A of GEMM1)
  __shared__ _Float16 Ts[ET][NFILT];   // ssp(hidden) (A of GEMM2)
  __shared__ float    Cs[ET];
  __shared__ float    Ds[ET];
  __shared__ int      Ss[ET];
  __shared__ int      Dd[ET];

  int e0 = blockIdx.x * ET;
  int t = threadIdx.x;

  if (t < ET) {
    int e = e0 + t;
    int s = (e < N_EDGES) ? src[e] : 0;
    int d = (e < N_EDGES) ? dst[e] : 0;
    Ss[t] = s; Dd[t] = d;
    float dx = pos[(size_t)s * 3 + 0] - pos[(size_t)d * 3 + 0];
    float dy = pos[(size_t)s * 3 + 1] - pos[(size_t)d * 3 + 1];
    float dz = pos[(size_t)s * 3 + 2] - pos[(size_t)d * 3 + 2];
    float dist = sqrtf(dx * dx + dy * dy + dz * dz);
    Ds[t] = dist;
    Cs[t] = 0.5f * (__cosf(dist * (PI_F / CUTOFF_F)) + 1.0f);
  }
  __syncthreads();

  // gaussian smearing into LDS (f16), zero padding for g >= 50
  for (int idx = t; idx < ET * NGP; idx += 256) {
    int el = idx >> 6, g = idx & (NGP - 1);
    float v = 0.0f;
    if (g < NG) { float diff = Ds[el] - (float)g * GSTEP; v = __expf(GCOEFF * diff * diff); }
    As[el][g] = (_Float16)v;
  }
  __syncthreads();

  int wave = t >> 5, lane = t & 31;
  int hf = lane >> 4, lr = lane & 15;
  int ncol = wave * 16 + lr;

  // GEMM1: [32 x 64] @ [64 x 128]
  v8f p0 = {}, p1 = {};
  for (int kb = 0; kb < NGP / 32; ++kb) {
    v16h b  = load_b_frag(Wm1, kb, wave, lane);
    v16h a0 = load_a_lds(&As[0][0], NGP, lr,      hf, kb * 32);
    v16h a1 = load_a_lds(&As[0][0], NGP, 16 + lr, hf, kb * 32);
    p0 = __builtin_amdgcn_wmma_f32_16x16x32_f16(false, a0, false, b, (short)0, p0, false, false);
    p1 = __builtin_amdgcn_wmma_f32_16x16x32_f16(false, a1, false, b, (short)0, p1, false, false);
  }
  float bias1 = bm1[ncol];
#pragma unroll
  for (int v = 0; v < 8; ++v) {
    int M = v + 8 * hf;
    Ts[M][ncol]      = (_Float16)sspf(p0[v] + bias1);
    Ts[16 + M][ncol] = (_Float16)sspf(p1[v] + bias1);
  }
  __syncthreads();

  // GEMM2: [32 x 128] @ [128 x 128]
  v8f q0 = {}, q1 = {};
  for (int kb = 0; kb < NFILT / 32; ++kb) {
    v16h b  = load_b_frag(Wm2, kb, wave, lane);
    v16h a0 = load_a_lds(&Ts[0][0], NFILT, lr,      hf, kb * 32);
    v16h a1 = load_a_lds(&Ts[0][0], NFILT, 16 + lr, hf, kb * 32);
    q0 = __builtin_amdgcn_wmma_f32_16x16x32_f16(false, a0, false, b, (short)0, q0, false, false);
    q1 = __builtin_amdgcn_wmma_f32_16x16x32_f16(false, a1, false, b, (short)0, q1, false, false);
  }
  float bias2 = bm2[ncol];
#pragma unroll
  for (int v = 0; v < 8; ++v) {
    int M = v + 8 * hf;
    {
      float Wfv = (q0[v] + bias2) * Cs[M];
      float msg = xf[(size_t)Ss[M] * NFILT + ncol] * Wfv;
      atomicAdd(&agg[(size_t)Dd[M] * NFILT + ncol], msg);
    }
    {
      int M1 = 16 + M;
      float Wfv = (q1[v] + bias2) * Cs[M1];
      float msg = xf[(size_t)Ss[M1] * NFILT + ncol] * Wfv;
      atomicAdd(&agg[(size_t)Dd[M1] * NFILT + ncol], msg);
    }
  }
}

// ---------- fused tail: xc = agg@W_cl2+b; h += ssp(xc)@W_lin+b (two chained WMMA GEMMs) ----------

__global__ __launch_bounds__(256) void tail_kernel(const float* __restrict__ agg,
                                                   const _Float16* __restrict__ Wc2, const float* __restrict__ bc2,
                                                   const _Float16* __restrict__ Wl,  const float* __restrict__ bl,
                                                   float* __restrict__ h) {
  __shared__ _Float16 As[NT][NFILT];
  __shared__ _Float16 Ts[NT][NFILT];
  int n0 = blockIdx.x * NT;
  int t = threadIdx.x;
  stage_rows_f16(agg + (size_t)n0 * NFILT, &As[0][0], t);
  __syncthreads();

  int wave = t >> 5, lane = t & 31;
  int hf = lane >> 4, lr = lane & 15;
  int ncol = wave * 16 + lr;

  v8f p0 = {}, p1 = {};
  for (int kb = 0; kb < NFILT / 32; ++kb) {
    v16h b  = load_b_frag(Wc2, kb, wave, lane);
    v16h a0 = load_a_lds(&As[0][0], NFILT, lr,      hf, kb * 32);
    v16h a1 = load_a_lds(&As[0][0], NFILT, 16 + lr, hf, kb * 32);
    p0 = __builtin_amdgcn_wmma_f32_16x16x32_f16(false, a0, false, b, (short)0, p0, false, false);
    p1 = __builtin_amdgcn_wmma_f32_16x16x32_f16(false, a1, false, b, (short)0, p1, false, false);
  }
  float bias1 = bc2[ncol];
#pragma unroll
  for (int v = 0; v < 8; ++v) {
    int M = v + 8 * hf;
    Ts[M][ncol]      = (_Float16)sspf(p0[v] + bias1);
    Ts[16 + M][ncol] = (_Float16)sspf(p1[v] + bias1);
  }
  __syncthreads();

  v8f q0 = {}, q1 = {};
  for (int kb = 0; kb < NFILT / 32; ++kb) {
    v16h b  = load_b_frag(Wl, kb, wave, lane);
    v16h a0 = load_a_lds(&Ts[0][0], NFILT, lr,      hf, kb * 32);
    v16h a1 = load_a_lds(&Ts[0][0], NFILT, 16 + lr, hf, kb * 32);
    q0 = __builtin_amdgcn_wmma_f32_16x16x32_f16(false, a0, false, b, (short)0, q0, false, false);
    q1 = __builtin_amdgcn_wmma_f32_16x16x32_f16(false, a1, false, b, (short)0, q1, false, false);
  }
  float bias2 = bl[ncol];
#pragma unroll
  for (int v = 0; v < 8; ++v) {
    int M = v + 8 * hf;
    h[(size_t)(n0 + M) * HID + ncol]      += q0[v] + bias2;
    h[(size_t)(n0 + 16 + M) * HID + ncol] += q1[v] + bias2;
  }
}

// ---------------- output head: ssp(h@W1+b1)@W2+b2, segment-sum by graph ----------------

__global__ __launch_bounds__(256) void head_kernel(const float* __restrict__ h,
                                                   const float* __restrict__ W1, const float* __restrict__ b1,
                                                   const float* __restrict__ W2, const float* __restrict__ b2,
                                                   const int* __restrict__ batch, float* __restrict__ out) {
  int wave = threadIdx.x >> 5, lane = threadIdx.x & 31;
  int n = blockIdx.x * 8 + wave;
  if (n >= N_NODES) return;
  float a0 = 0.0f, a1 = 0.0f;
  const float* hr = h + (size_t)n * HID;
  for (int k = 0; k < HID; ++k) {
    float hv = hr[k];
    a0 += hv * W1[(size_t)k * 64 + lane];
    a1 += hv * W1[(size_t)k * 64 + lane + 32];
  }
  float v = sspf(a0 + b1[lane]) * W2[lane] + sspf(a1 + b1[lane + 32]) * W2[lane + 32];
  for (int off = 16; off > 0; off >>= 1) v += __shfl_down(v, off, 32);
  if (lane == 0) atomicAdd(&out[batch[n]], v + b2[0]);
}

// ------------------------------------ launcher ------------------------------------

extern "C" void kernel_launch(void* const* d_in, const int* in_sizes, int n_in,
                              void* d_out, int out_size, void* d_ws, size_t ws_size,
                              hipStream_t stream) {
  const int*   z      = (const int*)d_in[0];
  const float* pos    = (const float*)d_in[1];
  const int*   eidx   = (const int*)d_in[2];
  const int*   batch  = (const int*)d_in[3];
  const float* emb    = (const float*)d_in[4];
  const float* W_mlp1 = (const float*)d_in[5];
  const float* b_mlp1 = (const float*)d_in[6];
  const float* W_mlp2 = (const float*)d_in[7];
  const float* b_mlp2 = (const float*)d_in[8];
  const float* W_cl1  = (const float*)d_in[9];
  const float* W_cl2  = (const float*)d_in[10];
  const float* b_cl2  = (const float*)d_in[11];
  const float* W_lin  = (const float*)d_in[12];
  const float* b_lin  = (const float*)d_in[13];
  const float* W_out1 = (const float*)d_in[14];
  const float* b_out1 = (const float*)d_in[15];
  const float* W_out2 = (const float*)d_in[16];
  const float* b_out2 = (const float*)d_in[17];
  float* out = (float*)d_out;

  const int* src = eidx;             // edge_index[0] = message source j
  const int* dst = eidx + N_EDGES;   // edge_index[1] = aggregation target i

  // workspace carve-up (~32 MB)
  char* ws = (char*)d_ws;
  float* h   = (float*)ws; ws += (size_t)N_NODES * HID * sizeof(float);
  float* xf  = (float*)ws; ws += (size_t)N_NODES * NFILT * sizeof(float);
  float* agg = (float*)ws; ws += (size_t)N_NODES * NFILT * sizeof(float);
  _Float16* m1_16 = (_Float16*)ws; ws += (size_t)NI * NGP * NFILT * sizeof(_Float16);
  _Float16* m2_16 = (_Float16*)ws; ws += (size_t)NI * NFILT * NFILT * sizeof(_Float16);
  _Float16* c1_16 = (_Float16*)ws; ws += (size_t)NI * NFILT * NFILT * sizeof(_Float16);
  _Float16* c2_16 = (_Float16*)ws; ws += (size_t)NI * NFILT * NFILT * sizeof(_Float16);
  _Float16* l_16  = (_Float16*)ws; ws += (size_t)NI * NFILT * NFILT * sizeof(_Float16);

  // zero output accumulator (graph replays re-zero every call: deterministic)
  zero_f32_kernel<<<1, 128, 0, stream>>>(out, N_GRAPHS);

  // convert weights to f16 fragment-ready layout (W_mlp1 padded 50->64 rows)
  {
    int total = NI * NGP * NFILT + 4 * NI * NFILT * NFILT;
    convert_weights_kernel<<<(total + 255) / 256, 256, 0, stream>>>(
        W_mlp1, W_mlp2, W_cl1, W_cl2, W_lin, m1_16, m2_16, c1_16, c2_16, l_16);
  }

  // h = emb[z]
  embed_kernel<<<(N_NODES * HID + 255) / 256, 256, 0, stream>>>(z, emb, h);

  const int nodeBlocks = N_NODES / NT;    // 625
  const int edgeBlocks = N_EDGES / ET;    // 20000

  for (int i = 0; i < NI; ++i) {
    lin1_kernel<<<nodeBlocks, 256, 0, stream>>>(h, c1_16 + (size_t)i * NFILT * NFILT, xf);
    zero_f32_kernel<<<1024, 256, 0, stream>>>(agg, N_NODES * NFILT);
    edge_kernel<<<edgeBlocks, 256, 0, stream>>>(
        pos, src, dst,
        m1_16 + (size_t)i * NGP * NFILT,   b_mlp1 + (size_t)i * NFILT,
        m2_16 + (size_t)i * NFILT * NFILT, b_mlp2 + (size_t)i * NFILT,
        xf, agg);
    tail_kernel<<<nodeBlocks, 256, 0, stream>>>(
        agg,
        c2_16 + (size_t)i * NFILT * NFILT, b_cl2 + (size_t)i * HID,
        l_16  + (size_t)i * NFILT * NFILT, b_lin + (size_t)i * HID,
        h);
  }

  head_kernel<<<(N_NODES + 7) / 8, 256, 0, stream>>>(h, W_out1, b_out1, W_out2, b_out2, batch, out);
}
